// HunyuanSplashAttnProcessor_12240656793743
// MI455X (gfx1250) — compile-verified, matrix operand order checked by reading, not run on hardware
//
#include <hip/hip_runtime.h>
#include <hip/hip_bf16.h>

typedef __bf16 bf16;
typedef __attribute__((ext_vector_type(8)))  float  v8f;
typedef __attribute__((ext_vector_type(16))) __bf16 v16bf;
typedef __attribute__((ext_vector_type(8)))  __bf16 v8bf;

// ---------------------------------------------------------------------------
// CDNA5 async global->LDS copy (ASYNCcnt), per CDNA5 ISA.
// Each active lane copies 16 bytes from its global address to its LDS address.
// ---------------------------------------------------------------------------
__device__ __forceinline__ unsigned lds_off(const void* p) {
  // generic pointer to LDS: addr[31:0] is the LDS byte address (aperture rule)
  return (unsigned)(unsigned long long)p;
}
__device__ __forceinline__ void async_b128(unsigned lds, const void* g) {
  asm volatile("global_load_async_to_lds_b128 %0, %1, off"
               :: "v"(lds), "v"((unsigned long long)g) : "memory");
}
__device__ __forceinline__ void wait_async0() {
  asm volatile("s_wait_asynccnt 0x0" ::: "memory");
}

// Build a 16-half WMMA A/B fragment from an LDS row (32-wide k window).
// Lanes 0-15 hold k {0..7,16..23}; lanes 16-31 hold k {8..15,24..31}.
__device__ __forceinline__ v16bf load_frag(const bf16* p, int hi) {
  v8bf lo = *(const v8bf*)(p + hi * 8);
  v8bf hi8 = *(const v8bf*)(p + 16 + hi * 8);
  v16bf r;
#pragma unroll
  for (int i = 0; i < 8; ++i) { r[i] = lo[i]; r[i + 8] = hi8[i]; }
  return r;
}

// ---------------------------------------------------------------------------
// fp32 -> bf16 elementwise conversion
// ---------------------------------------------------------------------------
__global__ void cvt_f32_bf16(const float* __restrict__ src, bf16* __restrict__ dst, long n) {
  long i = (long)blockIdx.x * blockDim.x + threadIdx.x;
  long stride = (long)gridDim.x * blockDim.x;
  for (; i < n; i += stride) dst[i] = (bf16)src[i];
}

// ---------------------------------------------------------------------------
// Tiled transpose + convert: dst[n][m] = (bf16)src[m][n]; src is MxN fp32.
// grid = (N/32, M/32), 256 threads.
// ---------------------------------------------------------------------------
__global__ __launch_bounds__(256)
void transpose_cvt_f32_bf16(const float* __restrict__ src, bf16* __restrict__ dst,
                            int M, int N) {
  __shared__ float t[32][33];
  const int bx = blockIdx.x << 5;  // n
  const int by = blockIdx.y << 5;  // m
  const int tx = threadIdx.x & 31, ty = threadIdx.x >> 5;
#pragma unroll
  for (int r = 0; r < 32; r += 8)
    t[ty + r][tx] = src[(long)(by + ty + r) * N + bx + tx];
  __syncthreads();
#pragma unroll
  for (int r = 0; r < 32; r += 8)
    dst[(long)(bx + ty + r) * M + by + tx] = (bf16)t[tx][ty + r];
}

// ---------------------------------------------------------------------------
// C(MxN) = A(MxK) @ Bt^T + bias ; A row-major bf16, Bt is B transposed (NxK)
// bf16, C fp32. 128x128 tile / block, 8 waves, double-buffered async staging.
// ---------------------------------------------------------------------------
#define GAS 40  // LDS row stride in halves (80B): conflict-free b128 frag reads
__global__ __launch_bounds__(256)
void wmma_gemm_bf16(const bf16* __restrict__ A, const bf16* __restrict__ Bt,
                    const float* __restrict__ bias, float* __restrict__ C,
                    int M, int N, int K) {
  __shared__ __align__(16) bf16 As[2][128 * GAS];
  __shared__ __align__(16) bf16 Bs[2][128 * GAS];
  const int tid = threadIdx.x, wave = tid >> 5, lane = tid & 31;
  const int nl = lane & 15, hi = lane >> 4;
  const int bm = blockIdx.y << 7, bn = blockIdx.x << 7;

  v8f acc[8] = {};

  auto kick = [&](int buf, int k0) {
#pragma unroll
    for (int c = tid; c < 512; c += 256) {
      int r = c >> 2, kc = (c & 3) << 3;  // 128 rows x 4 chunks of 8 halves
      async_b128(lds_off(&As[buf][r * GAS + kc]), &A[(long)(bm + r) * K + k0 + kc]);
      async_b128(lds_off(&Bs[buf][r * GAS + kc]), &Bt[(long)(bn + r) * K + k0 + kc]);
    }
  };

  const int nk = K >> 5;
  kick(0, 0);
  for (int ki = 0; ki < nk; ++ki) {
    wait_async0();
    __syncthreads();
    if (ki + 1 < nk) kick((ki + 1) & 1, (ki + 1) << 5);
    const int buf = ki & 1;
    v16bf af = load_frag(&As[buf][((wave << 4) + nl) * GAS], hi);
#pragma unroll
    for (int t = 0; t < 8; ++t) {
      v16bf bfr = load_frag(&Bs[buf][((t << 4) + nl) * GAS], hi);
      acc[t] = __builtin_amdgcn_wmma_f32_16x16x32_bf16(
          false, af, false, bfr, (short)0, acc[t], false, false);
    }
  }

#pragma unroll
  for (int t = 0; t < 8; ++t) {
    int col = bn + (t << 4) + nl;
    float bsv = bias[col];
#pragma unroll
    for (int r = 0; r < 8; ++r) {
      long row = bm + (wave << 4) + (hi << 3) + r;
      C[row * (long)N + col] = acc[t][r] + bsv;
    }
  }
}

// ---------------------------------------------------------------------------
// Per-head RMSNorm (+ RoPE for s < S_rope) + scale, fp32 -> bf16.
// One wave per (token, head) row of 128; 4 elements (2 rope pairs) per lane.
// ---------------------------------------------------------------------------
__global__ __launch_bounds__(256)
void norm_rope_kernel(const float* __restrict__ X, const float* __restrict__ w,
                      const float* __restrict__ cosv, const float* __restrict__ sinv,
                      bf16* __restrict__ Y, int S, int S_rope, float scale) {
  const int wave = threadIdx.x >> 5, lane = threadIdx.x & 31;
  long row = (long)blockIdx.x * 8 + wave;
  if (row >= (long)S * 24) return;
  int s = (int)(row / 24), h = (int)(row % 24);
  const int hd = lane << 2;
  const float* xp = X + (long)s * 3072 + h * 128 + hd;
  float v0 = xp[0], v1 = xp[1], v2 = xp[2], v3 = xp[3];
  float ss = v0 * v0 + v1 * v1 + v2 * v2 + v3 * v3;
#pragma unroll
  for (int m = 1; m < 32; m <<= 1) ss += __shfl_xor(ss, m, 32);
  float rms = rsqrtf(ss * (1.0f / 128.0f) + 1e-6f);
  v0 *= rms * w[hd]; v1 *= rms * w[hd + 1]; v2 *= rms * w[hd + 2]; v3 *= rms * w[hd + 3];
  if (s < S_rope) {
    const float* cp = cosv + (long)s * 128 + hd;
    const float* sp = sinv + (long)s * 128 + hd;
    float o0 = v0 * cp[0] - v1 * sp[0];
    float o1 = v1 * cp[1] + v0 * sp[1];
    float o2 = v2 * cp[2] - v3 * sp[2];
    float o3 = v3 * cp[3] + v2 * sp[3];
    v0 = o0; v1 = o1; v2 = o2; v3 = o3;
  }
  bf16* yp = Y + (long)s * 3072 + h * 128 + hd;
  yp[0] = (bf16)(v0 * scale); yp[1] = (bf16)(v1 * scale);
  yp[2] = (bf16)(v2 * scale); yp[3] = (bf16)(v3 * scale);
}

// ---------------------------------------------------------------------------
// Flash attention: 128 queries/block, double-buffered async K/V tiles of 64,
// online softmax in fp32. Q pre-scaled by 1/sqrt(128). V comes pre-transposed
// per head (Vtg: 3072 rows x 2304, row = h*128+d). Writes bf16 attn and fp32
// txt rows (>=2048) straight to d_out. grid = (18, 24), 256 threads.
// ---------------------------------------------------------------------------
#define QSTR 136
#define KSTR 136
#define VSTR 72
#define PSTR 72
__global__ __launch_bounds__(256)
void flash_attn_bf16(const bf16* __restrict__ Q, const bf16* __restrict__ Kt,
                     const bf16* __restrict__ Vtg, float* __restrict__ txt_out,
                     bf16* __restrict__ attn_b) {
  __shared__ __align__(16) bf16 Qs[128 * QSTR];
  __shared__ __align__(16) bf16 Ks[2][64 * KSTR];
  __shared__ __align__(16) bf16 Vt[2][128 * VSTR];  // [d][kv]
  __shared__ __align__(16) bf16 Ps[128 * PSTR];
  const int tid = threadIdx.x, wave = tid >> 5, lane = tid & 31;
  const int nl = lane & 15, hi = lane >> 4;
  const int h = blockIdx.y;
  const int q0 = blockIdx.x << 7;
  const int colbase = h << 7;

  // Q tile 128x128 (async)
#pragma unroll
  for (int c = tid; c < 2048; c += 256) {
    int r = c >> 4, kc = (c & 15) << 3;
    async_b128(lds_off(&Qs[r * QSTR + kc]), &Q[(long)(q0 + r) * 3072 + colbase + kc]);
  }

  auto kick_kv = [&](int buf, int kv0) {
#pragma unroll
    for (int c = tid; c < 1024; c += 256) {  // K: 64 rows x 16 chunks
      int r = c >> 4, kc = (c & 15) << 3;
      async_b128(lds_off(&Ks[buf][r * KSTR + kc]),
                 &Kt[(long)(kv0 + r) * 3072 + colbase + kc]);
    }
#pragma unroll
    for (int c = tid; c < 1024; c += 256) {  // V: 128 d-rows x 8 chunks of 8 kv
      int d = c >> 3, kc = (c & 7) << 3;
      async_b128(lds_off(&Vt[buf][d * VSTR + kc]),
                 &Vtg[(long)(colbase + d) * 2304 + kv0 + kc]);
    }
  };
  kick_kv(0, 0);

  float mrow[8], lrow[8];
  v8f oacc[8] = {};
#pragma unroll
  for (int r = 0; r < 8; ++r) { mrow[r] = -1e30f; lrow[r] = 0.0f; }

  for (int kt = 0; kt < 36; ++kt) {
    wait_async0();
    __syncthreads();
    if (kt + 1 < 36) kick_kv((kt + 1) & 1, (kt + 1) << 6);
    const int buf = kt & 1;

    // S = Q @ K^T : 4 k-steps of 32 over HD, 4 n-tiles of 16 over 64 keys
    v8f sacc[4] = {};
#pragma unroll
    for (int kk = 0; kk < 4; ++kk) {
      v16bf af = load_frag(&Qs[((wave << 4) + nl) * QSTR + (kk << 5)], hi);
#pragma unroll
      for (int t = 0; t < 4; ++t) {
        v16bf bfr = load_frag(&Ks[buf][((t << 4) + nl) * KSTR + (kk << 5)], hi);
        sacc[t] = __builtin_amdgcn_wmma_f32_16x16x32_bf16(
            false, af, false, bfr, (short)0, sacc[t], false, false);
      }
    }

    // Online softmax; rows live in 16-lane halves per the C-matrix layout.
#pragma unroll
    for (int r = 0; r < 8; ++r) {
      float mx = fmaxf(fmaxf(sacc[0][r], sacc[1][r]), fmaxf(sacc[2][r], sacc[3][r]));
      mx = fmaxf(mx, __shfl_xor(mx, 1, 32));
      mx = fmaxf(mx, __shfl_xor(mx, 2, 32));
      mx = fmaxf(mx, __shfl_xor(mx, 4, 32));
      mx = fmaxf(mx, __shfl_xor(mx, 8, 32));
      float mn = fmaxf(mrow[r], mx);
      float alpha = __expf(mrow[r] - mn);
      mrow[r] = mn;
      float sum = 0.0f;
#pragma unroll
      for (int t = 0; t < 4; ++t) { float p = __expf(sacc[t][r] - mn); sacc[t][r] = p; sum += p; }
      sum += __shfl_xor(sum, 1, 32);
      sum += __shfl_xor(sum, 2, 32);
      sum += __shfl_xor(sum, 4, 32);
      sum += __shfl_xor(sum, 8, 32);
      lrow[r] = lrow[r] * alpha + sum;
#pragma unroll
      for (int dd = 0; dd < 8; ++dd) oacc[dd][r] *= alpha;
    }

    // P: C-layout -> A-layout via LDS (same-wave rows only; LDS is in-order)
#pragma unroll
    for (int t = 0; t < 4; ++t)
#pragma unroll
      for (int r = 0; r < 8; ++r)
        Ps[((wave << 4) + (hi << 3) + r) * PSTR + (t << 4) + nl] = (bf16)sacc[t][r];

    // O += P @ V : 2 k-steps of 32 over 64 keys, 8 d-tiles of 16
    v16bf ap0 = load_frag(&Ps[((wave << 4) + nl) * PSTR + 0], hi);
    v16bf ap1 = load_frag(&Ps[((wave << 4) + nl) * PSTR + 32], hi);
#pragma unroll
    for (int dd = 0; dd < 8; ++dd) {
      v16bf b0 = load_frag(&Vt[buf][((dd << 4) + nl) * VSTR + 0], hi);
      oacc[dd] = __builtin_amdgcn_wmma_f32_16x16x32_bf16(
          false, ap0, false, b0, (short)0, oacc[dd], false, false);
      v16bf b1 = load_frag(&Vt[buf][((dd << 4) + nl) * VSTR + 32], hi);
      oacc[dd] = __builtin_amdgcn_wmma_f32_16x16x32_bf16(
          false, ap1, false, b1, (short)0, oacc[dd], false, false);
    }
  }

#pragma unroll
  for (int r = 0; r < 8; ++r) {
    float inv = 1.0f / lrow[r];
    long srow = q0 + (wave << 4) + (hi << 3) + r;
#pragma unroll
    for (int dd = 0; dd < 8; ++dd) {
      float o = oacc[dd][r] * inv;
      long col = colbase + (dd << 4) + nl;
      attn_b[srow * 3072 + col] = (bf16)o;
      if (srow >= 2048) txt_out[(srow - 2048) * 3072 + col] = o;
    }
  }
}

// ---------------------------------------------------------------------------
extern "C" void kernel_launch(void* const* d_in, const int* in_sizes, int n_in,
                              void* d_out, int out_size, void* d_ws, size_t ws_size,
                              hipStream_t stream) {
  const float* hs  = (const float*)d_in[0];
  const float* ehs = (const float*)d_in[1];
  const float* rc  = (const float*)d_in[2];
  const float* rs  = (const float*)d_in[3];
  const float* Wq  = (const float*)d_in[4];
  const float* bq  = (const float*)d_in[5];
  const float* Wk  = (const float*)d_in[6];
  const float* bk  = (const float*)d_in[7];
  const float* Wv  = (const float*)d_in[8];
  const float* bv  = (const float*)d_in[9];
  const float* nqw = (const float*)d_in[10];
  const float* nkw = (const float*)d_in[11];
  const float* Wo  = (const float*)d_in[12];
  const float* bo  = (const float*)d_in[13];

  const long S = 2304, SIMG = 2048, STXT = 256, D = 3072;
  char* ws = (char*)d_ws;
  size_t off = 0;
  auto take = [&](size_t bytes) -> char* {
    char* p = ws + off; off = (off + bytes + 255) & ~(size_t)255; return p;
  };
  bf16*  Ax   = (bf16*)take(S * D * 2);
  bf16*  Wqt  = (bf16*)take(D * D * 2);   // W^T, N x K bf16
  bf16*  Wkt  = (bf16*)take(D * D * 2);
  bf16*  Wvt  = (bf16*)take(D * D * 2);
  float* Qraw = (float*)take(S * D * 4);
  float* Kraw = (float*)take(S * D * 4);
  float* Vraw = (float*)take(S * D * 4);
  bf16*  Qn   = (bf16*)take(S * D * 2);
  bf16*  Kn   = (bf16*)take(S * D * 2);
  bf16*  Vtg  = (bf16*)take(D * S * 2);   // V^T per head: (h*128+d) x S
  bf16*  attn_b = Ax;          // Ax dead after QKV GEMMs
  bf16*  Wot    = (bf16*)Qraw; // Qraw dead after norm_rope(Q)

  // x = concat(img, txt) -> bf16 ; weights -> transposed bf16 (NxK)
  cvt_f32_bf16<<<2048, 256, 0, stream>>>(hs,  Ax,            SIMG * D);
  cvt_f32_bf16<<<2048, 256, 0, stream>>>(ehs, Ax + SIMG * D, STXT * D);
  dim3 gtw(96, 96);
  transpose_cvt_f32_bf16<<<gtw, 256, 0, stream>>>(Wq, Wqt, 3072, 3072);
  transpose_cvt_f32_bf16<<<gtw, 256, 0, stream>>>(Wk, Wkt, 3072, 3072);
  transpose_cvt_f32_bf16<<<gtw, 256, 0, stream>>>(Wv, Wvt, 3072, 3072);

  // QKV projections
  dim3 gq(24, 18);
  wmma_gemm_bf16<<<gq, 256, 0, stream>>>(Ax, Wqt, bq, Qraw, 2304, 3072, 3072);
  wmma_gemm_bf16<<<gq, 256, 0, stream>>>(Ax, Wkt, bk, Kraw, 2304, 3072, 3072);
  wmma_gemm_bf16<<<gq, 256, 0, stream>>>(Ax, Wvt, bv, Vraw, 2304, 3072, 3072);

  // RMSNorm + RoPE (1/sqrt(HD) folded into Q); V -> per-head transposed bf16
  const float scale = 0.08838834764831845f;  // 1/sqrt(128)
  norm_rope_kernel<<<6912, 256, 0, stream>>>(Qraw, nqw, rc, rs, Qn, 2304, 2048, scale);
  norm_rope_kernel<<<6912, 256, 0, stream>>>(Kraw, nkw, rc, rs, Kn, 2304, 2048, 1.0f);
  dim3 gtv(96, 72);
  transpose_cvt_f32_bf16<<<gtv, 256, 0, stream>>>(Vraw, Vtg, 2304, 3072);
  transpose_cvt_f32_bf16<<<gtw, 256, 0, stream>>>(Wo, Wot, 3072, 3072);  // into dead Qraw

  // Attention (txt rows stream straight to d_out)
  dim3 ga(18, 24);
  flash_attn_bf16<<<ga, 256, 0, stream>>>(Qn, Kn, Vtg, (float*)d_out + SIMG * D, attn_b);

  // img_out = attn[:2048] @ Wo + bo
  dim3 go(24, 16);
  wmma_gemm_bf16<<<go, 256, 0, stream>>>(attn_b, Wot, bo, (float*)d_out, 2048, 3072, 3072);
}
